// TransformerProgressHead_49091476193566
// MI455X (gfx1250) — matrix operand order repaired and checked by reading, not compiled, not run on hardware
//
#include <hip/hip_runtime.h>
#include <hip/hip_bf16.h>
#include <math.h>

#define T_ 2048
#define D_ 1024
#define H_ 16
#define HD_ 64
#define F_ 4096
#define L_ 4

typedef _Float16 hv8  __attribute__((ext_vector_type(8)));
typedef _Float16 hv16 __attribute__((ext_vector_type(16)));
typedef float    fv8  __attribute__((ext_vector_type(8)));

// ---------------------------------------------------------------------------
// WMMA fragment loaders (wave32; A and B both use contiguous K order)
// A 16x32 f16: lane holds row m=lane&15, K = hi*16 .. hi*16+15 contiguous
// B 32x16 f16 from row-major B^T (rows=N): lane holds col n=lane&15,
//   K = hi*16 .. hi*16+15 contiguous.
// C/D 16x16 f32: VGPR r, lane l: M = r + (l>=16)*8, N = l&15
// ---------------------------------------------------------------------------
__device__ __forceinline__ hv16 load_a_frag(const _Float16* base, int ld) {
  int lane = threadIdx.x & 31;
  int m = lane & 15, hi = lane >> 4;
  return *(const hv16*)(base + (size_t)m * ld + hi * 16);
}

__device__ __forceinline__ hv16 load_b_frag(const _Float16* baseT, int ld) {
  int lane = threadIdx.x & 31;
  int n = lane & 15, hi = lane >> 4;
  return *(const hv16*)(baseT + (size_t)n * ld + hi * 16);
}

__device__ __forceinline__ fv8 wmma_f16(hv16 a, hv16 b, fv8 c) {
  return __builtin_amdgcn_wmma_f32_16x16x32_f16(false, a, false, b, (short)0, c,
                                                false, false);
}

// CDNA5 async global->LDS copy (ASYNCcnt path); GV mode, VDST = LDS byte addr.
__device__ __forceinline__ void async_copy_b128(unsigned int lds_addr,
                                                const void* gaddr) {
  asm volatile("global_load_async_to_lds_b128 %0, %1, off"
               :
               : "v"(lds_addr), "v"((unsigned long long)(uintptr_t)gaddr)
               : "memory");
}
__device__ __forceinline__ void wait_async0() {
  asm volatile("s_wait_asynccnt 0" ::: "memory");
}

// ---------------------------------------------------------------------------
// WMMA GEMM: C[M,N] = A[MxK f16] * B (given as B^T, NxK f16) + bias
// flags: 1 = exact GELU, 2 = add residual (f32 MxN)
// Outputs (any may be null): outF (f32 MxN), outH (f16 MxN), outHT (f16 NxM)
// grid = (M/32, N/128), block = 128 (4 waves). Block tile 32M x 128N;
// wave tile 32M x 32N (4 WMMAs / K-step). The 32x32 A tile is staged in LDS
// with async-to-LDS loads (2 KB/step, 16 B per thread), shared by all waves.
// ---------------------------------------------------------------------------
__global__ void gemm_wmma_kernel(const _Float16* __restrict__ A,
                                 const _Float16* __restrict__ Bt,
                                 const float* __restrict__ bias,
                                 const float* __restrict__ residual,
                                 float* __restrict__ outF,
                                 _Float16* __restrict__ outH,
                                 _Float16* __restrict__ outHT,
                                 int M, int N, int K, int flags) {
  __shared__ _Float16 ash[32 * 32];  // 32 rows x 32 K-halfs
  int tid = threadIdx.x;
  int wave = tid >> 5;
  int lane = tid & 31;
  int mBase = blockIdx.x * 32;
  int nBase = blockIdx.y * 128 + wave * 32;

  // per-thread async-copy slot: row = tid/4 (0..31), chunk = tid%4 (8 halfs)
  int arow = tid >> 2, achk = tid & 3;
  unsigned int lds_dst =
      (unsigned int)(uintptr_t)(&ash[0]) + (unsigned int)(arow * 64 + achk * 16);
  const char* gsrc =
      (const char*)(A + (size_t)(mBase + arow) * K + achk * 8);

  fv8 c00 = {}, c01 = {}, c10 = {}, c11 = {};
  const _Float16* Bt0 = Bt + (size_t)nBase * K;
  const _Float16* Bt1 = Bt + (size_t)(nBase + 16) * K;

  for (int k = 0; k < K; k += 32) {
    async_copy_b128(lds_dst, gsrc + (size_t)k * 2);
    wait_async0();
    __syncthreads();
    hv16 a0 = load_a_frag(ash, 32);            // rows mBase..mBase+15
    hv16 a1 = load_a_frag(ash + 16 * 32, 32);  // rows mBase+16..mBase+31
    hv16 b0 = load_b_frag(Bt0 + k, K);
    hv16 b1 = load_b_frag(Bt1 + k, K);
    c00 = wmma_f16(a0, b0, c00);
    c01 = wmma_f16(a0, b1, c01);
    c10 = wmma_f16(a1, b0, c10);
    c11 = wmma_f16(a1, b1, c11);
    __syncthreads();
  }

  int hi = lane >> 4, nn = lane & 15;
#pragma unroll
  for (int mt = 0; mt < 2; ++mt) {
#pragma unroll
    for (int r = 0; r < 8; ++r) {
      int m = mBase + mt * 16 + r + hi * 8;
#pragma unroll
      for (int t = 0; t < 2; ++t) {
        int n = nBase + t * 16 + nn;
        float v;
        if (mt == 0) v = (t == 0 ? c00[r] : c01[r]);
        else         v = (t == 0 ? c10[r] : c11[r]);
        v += bias[n];
        if (flags & 1) v = 0.5f * v * (1.0f + erff(v * 0.70710678118654752f));
        if (flags & 2) v += residual[(size_t)m * N + n];
        if (outF)  outF[(size_t)m * N + n] = v;
        if (outH)  outH[(size_t)m * N + n] = (_Float16)v;
        if (outHT) outHT[(size_t)n * M + m] = (_Float16)v;
      }
    }
  }
}

// ---------------------------------------------------------------------------
// Flash-style causal attention with ALiBi. One wave per (16-query block, head)
// Q,K: f16 TxD row-major.  Vt: f16 DxT (V transposed).  O16: f16 TxD.
// ---------------------------------------------------------------------------
__global__ void attn_kernel(const _Float16* __restrict__ Q,
                            const _Float16* __restrict__ Km,
                            const _Float16* __restrict__ Vt,
                            _Float16* __restrict__ O16) {
  __shared__ _Float16 pbuf[16 * 32];
  int qb = blockIdx.x, h = blockIdx.y;
  int lane = threadIdx.x & 31;
  int hi = lane >> 4, nn = lane & 15;
  int qbase = qb * 16;
  const float slope = exp2f(-(float)h);

  hv16 qa0 = load_a_frag(Q + (size_t)qbase * D_ + h * HD_, D_);
  hv16 qa1 = load_a_frag(Q + (size_t)qbase * D_ + h * HD_ + 32, D_);

  fv8 o[4]; o[0] = fv8{}; o[1] = fv8{}; o[2] = fv8{}; o[3] = fv8{};
  float m_[8], l_[8];
#pragma unroll
  for (int r = 0; r < 8; ++r) { m_[r] = -1e30f; l_[r] = 0.0f; }

  int nkb = (qbase + 16 + 31) >> 5;  // 32-key blocks covering keys <= qbase+15
  for (int kb = 0; kb < nkb; ++kb) {
    int kbase = kb * 32;
    // scores: S = Q(16x64) * K^T -> two 16x16 tiles
    fv8 s0 = {}, s1 = {};
    {
      const _Float16* k0 = Km + (size_t)kbase * D_ + h * HD_;
      s0 = wmma_f16(qa0, load_b_frag(k0, D_), s0);
      s0 = wmma_f16(qa1, load_b_frag(k0 + 32, D_), s0);
      const _Float16* k1 = Km + (size_t)(kbase + 16) * D_ + h * HD_;
      s1 = wmma_f16(qa0, load_b_frag(k1, D_), s1);
      s1 = wmma_f16(qa1, load_b_frag(k1 + 32, D_), s1);
    }
#pragma unroll
    for (int r = 0; r < 8; ++r) {
      int i = qbase + r + hi * 8;
      int j0 = kbase + nn;
      int j1 = kbase + 16 + nn;
      float a0 = s0[r] * 0.125f - slope * (float)(i - j0);
      float a1 = s1[r] * 0.125f - slope * (float)(i - j1);
      a0 = (j0 > i) ? -1e30f : a0;
      a1 = (j1 > i) ? -1e30f : a1;
      // row max across the 16 lanes holding this row
      float t = fmaxf(a0, a1);
#pragma unroll
      for (int mk = 8; mk >= 1; mk >>= 1) t = fmaxf(t, __shfl_xor(t, mk, 32));
      float newm = fmaxf(m_[r], t);
      float corr = expf(m_[r] - newm);
      m_[r] = newm;
      float p0 = expf(a0 - newm);
      float p1 = expf(a1 - newm);
      float ps = p0 + p1;
#pragma unroll
      for (int mk = 8; mk >= 1; mk >>= 1) ps += __shfl_xor(ps, mk, 32);
      l_[r] = l_[r] * corr + ps;
      o[0][r] *= corr; o[1][r] *= corr; o[2][r] *= corr; o[3][r] *= corr;
      int row = r + hi * 8;
      pbuf[row * 32 + nn]      = (_Float16)p0;
      pbuf[row * 32 + 16 + nn] = (_Float16)p1;
    }
    // P (16x32) back in A-fragment layout (same-wave DS ops are in-order)
    hv16 pa = load_a_frag(pbuf, 32);
#pragma unroll
    for (int t = 0; t < 4; ++t) {
      const _Float16* vb = Vt + (size_t)(h * HD_ + t * 16) * T_ + kbase;
      o[t] = wmma_f16(pa, load_b_frag(vb, T_), o[t]);
    }
  }
#pragma unroll
  for (int r = 0; r < 8; ++r) {
    float inv = 1.0f / l_[r];
    int row = qbase + r + hi * 8;
#pragma unroll
    for (int t = 0; t < 4; ++t)
      O16[(size_t)row * D_ + h * HD_ + t * 16 + nn] = (_Float16)(o[t][r] * inv);
  }
}

// ---------------------------------------------------------------------------
// LayerNorm over rows of in (rows x Dd). Optional f32 / f16 outputs.
// ---------------------------------------------------------------------------
__global__ void ln_kernel(const float* __restrict__ in,
                          const float* __restrict__ g,
                          const float* __restrict__ b,
                          float* __restrict__ outF,
                          _Float16* __restrict__ outH, int Dd) {
  __shared__ float red[256];
  int row = blockIdx.x, tid = threadIdx.x;
  const float* x = in + (size_t)row * Dd;
  float s = 0.0f;
  for (int i = tid; i < Dd; i += 256) s += x[i];
  red[tid] = s; __syncthreads();
  for (int off = 128; off > 0; off >>= 1) {
    if (tid < off) red[tid] += red[tid + off];
    __syncthreads();
  }
  float mu = red[0] / Dd; __syncthreads();
  float v = 0.0f;
  for (int i = tid; i < Dd; i += 256) { float d = x[i] - mu; v += d * d; }
  red[tid] = v; __syncthreads();
  for (int off = 128; off > 0; off >>= 1) {
    if (tid < off) red[tid] += red[tid + off];
    __syncthreads();
  }
  float rstd = rsqrtf(red[0] / Dd + 1e-5f);
  for (int i = tid; i < Dd; i += 256) {
    float y = (x[i] - mu) * rstd * g[i] + b[i];
    if (outF) outF[(size_t)row * Dd + i] = y;
    if (outH) outH[(size_t)row * Dd + i] = (_Float16)y;
  }
}

// ---------------------------------------------------------------------------
// Transpose f32 KxN -> f16 NxK (tiled through LDS). Dims multiples of 32.
// ---------------------------------------------------------------------------
__global__ void transpose_f16_kernel(const float* __restrict__ in,
                                     _Float16* __restrict__ out, int K, int N) {
  __shared__ float tile[32][33];
  int nb = blockIdx.x * 32, kb = blockIdx.y * 32;
  int tx = threadIdx.x, ty = threadIdx.y;  // block (32,8)
  for (int i = ty; i < 32; i += 8)
    tile[i][tx] = in[(size_t)(kb + i) * N + nb + tx];
  __syncthreads();
  for (int i = ty; i < 32; i += 8)
    out[(size_t)(nb + i) * K + kb + tx] = (_Float16)tile[tx][i];
}

__global__ void conv_f16_kernel(const float* __restrict__ in,
                                _Float16* __restrict__ out, int n) {
  int i = blockIdx.x * 256 + threadIdx.x;
  if (i < n) out[i] = (_Float16)in[i];
}

// bias2[d] = in_b[d] + fc * in_w[1024*D + d]
__global__ void bias2_kernel(const float* __restrict__ in_w,
                             const float* __restrict__ in_b, float fc,
                             float* __restrict__ bias2) {
  int d = blockIdx.x * 256 + threadIdx.x;
  if (d < D_) bias2[d] = in_b[d] + fc * in_w[(size_t)1024 * D_ + d];
}

// progress head: relu(x[-1] @ Wm1 + bm1) @ Wm2 + bm2 -> sigmoid
__global__ void head_kernel(const float* __restrict__ x,
                            const float* __restrict__ Wm1,
                            const float* __restrict__ bm1,
                            const float* __restrict__ Wm2,
                            const float* __restrict__ bm2,
                            float* __restrict__ out) {
  const float* fin = x + (size_t)(T_ - 1) * D_;
  int j = threadIdx.x;  // 0..31
  float acc = bm1[j];
  for (int i = 0; i < D_; ++i) acc += fin[i] * Wm1[(size_t)i * 32 + j];
  acc = fmaxf(acc, 0.0f);
  float p = acc * Wm2[j];
#pragma unroll
  for (int mk = 16; mk >= 1; mk >>= 1) p += __shfl_xor(p, mk, 32);
  if (j == 0) out[0] = 1.0f / (1.0f + expf(-(p + bm2[0])));
}

// ---------------------------------------------------------------------------
extern "C" void kernel_launch(void* const* d_in, const int* in_sizes, int n_in,
                              void* d_out, int out_size, void* d_ws,
                              size_t ws_size, hipStream_t stream) {
  (void)in_sizes; (void)n_in; (void)out_size; (void)ws_size;
  const float* seg    = (const float*)d_in[0];
  const float* in_w   = (const float*)d_in[1];
  const float* in_b   = (const float*)d_in[2];
  const float* lng    = (const float*)d_in[3];
  const float* lnb    = (const float*)d_in[4];
  const float* Wq     = (const float*)d_in[5];
  const float* bq     = (const float*)d_in[6];
  const float* Wk     = (const float*)d_in[7];
  const float* bk     = (const float*)d_in[8];
  const float* Wv     = (const float*)d_in[9];
  const float* bv     = (const float*)d_in[10];
  const float* Wo     = (const float*)d_in[11];
  const float* bo     = (const float*)d_in[12];
  const float* n1g    = (const float*)d_in[13];
  const float* n1b    = (const float*)d_in[14];
  const float* n2g    = (const float*)d_in[15];
  const float* n2b    = (const float*)d_in[16];
  const float* W1     = (const float*)d_in[17];
  const float* b1     = (const float*)d_in[18];
  const float* W2     = (const float*)d_in[19];
  const float* b2     = (const float*)d_in[20];
  const float* Wm1    = (const float*)d_in[21];
  const float* bm1    = (const float*)d_in[22];
  const float* Wm2    = (const float*)d_in[23];
  const float* bm2    = (const float*)d_in[24];

  // workspace carve
  char* p = (char*)d_ws;
  auto carve = [&](size_t bytes) {
    void* r = (void*)p;
    p += (bytes + 255) & ~(size_t)255;
    return r;
  };
  float*    x     = (float*)   carve((size_t)T_ * D_ * 4);
  float*    tmpf  = (float*)   carve((size_t)T_ * D_ * 4);
  float*    bias2 = (float*)   carve((size_t)D_ * 4);
  _Float16* xn16  = (_Float16*)carve((size_t)T_ * D_ * 2);
  _Float16* q16   = (_Float16*)carve((size_t)T_ * D_ * 2);
  _Float16* k16   = (_Float16*)carve((size_t)T_ * D_ * 2);
  _Float16* vt16  = (_Float16*)carve((size_t)D_ * T_ * 2);
  _Float16* ao16  = (_Float16*)carve((size_t)T_ * D_ * 2);
  _Float16* hbuf  = (_Float16*)carve((size_t)T_ * F_ * 2);
  _Float16* inwt  = (_Float16*)carve((size_t)D_ * D_ * 2);
  _Float16* wqt   = (_Float16*)carve((size_t)D_ * D_ * 2);
  _Float16* wkt   = (_Float16*)carve((size_t)D_ * D_ * 2);
  _Float16* wvt   = (_Float16*)carve((size_t)D_ * D_ * 2);
  _Float16* wot   = (_Float16*)carve((size_t)D_ * D_ * 2);
  _Float16* w1t   = (_Float16*)carve((size_t)D_ * F_ * 2);
  _Float16* w2t   = (_Float16*)carve((size_t)F_ * D_ * 2);

  const float fc = log1pf((float)T_) / log1pf(300.0f);

  dim3 trQ(D_ / 32, D_ / 32), trB(32, 8);
  dim3 g1(T_ / 32, D_ / 128);  // GEMM N=D
  dim3 gF(T_ / 32, F_ / 128);  // GEMM N=F

  // input projection + LN
  bias2_kernel<<<(D_ + 255) / 256, 256, 0, stream>>>(in_w, in_b, fc, bias2);
  conv_f16_kernel<<<(T_ * D_ + 255) / 256, 256, 0, stream>>>(seg, xn16, T_ * D_);
  transpose_f16_kernel<<<trQ, trB, 0, stream>>>(in_w, inwt, 1024, D_);
  gemm_wmma_kernel<<<g1, 128, 0, stream>>>(xn16, inwt, bias2, nullptr, tmpf,
                                           nullptr, nullptr, T_, D_, 1024, 0);
  ln_kernel<<<T_, 256, 0, stream>>>(tmpf, lng, lnb, x, nullptr, D_);

  for (int l = 0; l < L_; ++l) {
    const size_t dd = (size_t)l * D_ * D_;
    // pre-attention LN -> f16
    ln_kernel<<<T_, 256, 0, stream>>>(x, n1g + l * D_, n1b + l * D_, nullptr,
                                      xn16, D_);
    // weight transposes (f32 -> f16, N x K)
    transpose_f16_kernel<<<trQ, trB, 0, stream>>>(Wq + dd, wqt, D_, D_);
    transpose_f16_kernel<<<trQ, trB, 0, stream>>>(Wk + dd, wkt, D_, D_);
    transpose_f16_kernel<<<trQ, trB, 0, stream>>>(Wv + dd, wvt, D_, D_);
    transpose_f16_kernel<<<trQ, trB, 0, stream>>>(Wo + dd, wot, D_, D_);
    // Q, K, V^T
    gemm_wmma_kernel<<<g1, 128, 0, stream>>>(xn16, wqt, bq + l * D_, nullptr,
                                             nullptr, q16, nullptr, T_, D_, D_, 0);
    gemm_wmma_kernel<<<g1, 128, 0, stream>>>(xn16, wkt, bk + l * D_, nullptr,
                                             nullptr, k16, nullptr, T_, D_, D_, 0);
    gemm_wmma_kernel<<<g1, 128, 0, stream>>>(xn16, wvt, bv + l * D_, nullptr,
                                             nullptr, nullptr, vt16, T_, D_, D_, 0);
    // attention
    attn_kernel<<<dim3(T_ / 16, H_), 32, 0, stream>>>(q16, k16, vt16, ao16);
    // output projection + residual (in-place into x)
    gemm_wmma_kernel<<<g1, 128, 0, stream>>>(ao16, wot, bo + l * D_, x, x,
                                             nullptr, nullptr, T_, D_, D_, 2);
    // FFN
    ln_kernel<<<T_, 256, 0, stream>>>(x, n2g + l * D_, n2b + l * D_, nullptr,
                                      xn16, D_);
    transpose_f16_kernel<<<dim3(F_ / 32, D_ / 32), trB, 0, stream>>>(
        W1 + (size_t)l * D_ * F_, w1t, D_, F_);
    gemm_wmma_kernel<<<gF, 128, 0, stream>>>(xn16, w1t, b1 + l * F_, nullptr,
                                             nullptr, hbuf, nullptr, T_, F_, D_,
                                             1 /*gelu*/);
    transpose_f16_kernel<<<dim3(D_ / 32, F_ / 32), trB, 0, stream>>>(
        W2 + (size_t)l * F_ * D_, w2t, F_, D_);
    gemm_wmma_kernel<<<g1, 128, 0, stream>>>(hbuf, w2t, b2 + l * D_, x, x,
                                             nullptr, nullptr, T_, D_, F_,
                                             2 /*residual*/);
  }

  head_kernel<<<1, 32, 0, stream>>>(x, Wm1, bm1, Wm2, bm2, (float*)d_out);
}
